// SegNet_23742579212941
// MI455X (gfx1250) — compile-verified, matrix operand order checked by reading, not run on hardware
//
#include <hip/hip_runtime.h>
#include <hip/hip_bf16.h>

typedef __attribute__((ext_vector_type(16))) _Float16 v16h;
typedef __attribute__((ext_vector_type(8)))  _Float16 v8h;
typedef __attribute__((ext_vector_type(8)))  float    v8f;

#define BN_EPS 1e-5f

// ---------------------------------------------------------------------------
// Weight packing: fp32 OIHW -> sign-binarized f16, pre-swizzled into WMMA
// A-fragment order. Layout: wp[tile][chunk][lane][e] where
//   tile  = cout/16, chunk = cc*9 + tap  (channel-slab-major: A streams
//   sequentially inside the unrolled tap loop), lane 0..31, e 0..15.
// Fragment mapping: m = lane%16 (cout), koff(e,hi) = (e<8?e:e+8)+8*(lane>=16).
// Each lane's fragment = 32 contiguous bytes -> 2x global_load_b128.
// ---------------------------------------------------------------------------
__global__ void pack_weights(const float* __restrict__ w, _Float16* __restrict__ wp,
                             int Cin, int Cout, int CinPad, int CoutPad) {
  int nChunks = 9 * (CinPad >> 5);
  int total = (CoutPad >> 4) * nChunks * 512;   // 32 lanes * 16 elems
  int idx = blockIdx.x * blockDim.x + threadIdx.x;
  if (idx >= total) return;
  int e     = idx & 15;
  int lane  = (idx >> 4) & 31;
  int chunk = (idx >> 9) % nChunks;
  int tile  = (idx >> 9) / nChunks;
  int cc    = chunk / 9;
  int tap   = chunk - cc * 9;
  int hi    = lane >> 4;
  int koff  = ((e < 8) ? e : (e + 8)) + (hi << 3);
  int c     = (cc << 5) + koff;
  int cout  = (tile << 4) + (lane & 15);
  float val = 0.f;
  if (cout < Cout && c < Cin) {
    float ww = w[((size_t)cout * Cin + c) * 9 + tap];   // OIHW, tap = ky*3+kx
    val = (ww > 0.f) ? 1.f : ((ww < 0.f) ? -1.f : 0.f); // jnp.sign semantics
  }
  wp[idx] = (_Float16)val;
}

__global__ void f32_to_f16(const float* __restrict__ in, _Float16* __restrict__ out, size_t n) {
  size_t i = (size_t)blockIdx.x * blockDim.x + threadIdx.x;
  if (i < n) out[i] = (_Float16)in[i];
}

// ---------------------------------------------------------------------------
// Implicit-GEMM 3x3 SAME conv via v_wmma_f32_16x16x32_f16, LDS-staged B.
//
// Per 32-channel slab, each wave fills a PRIVATE LDS tile transposed to
// [pixel][channel]: 3 rows x 18 cols (16-pixel strip + 1-halo), 32 ch each.
// Fill is vectorized: each lane (= one channel) loads its row segment as
// 2x v8h (b128) + 2 clamped edge u16, then scatters 18 b16 LDS stores with
// immediate offsets; halo zeroing via cndmask on clamped loads (no saveexec).
// One fill serves all 9 taps; per tap the B fragment is 2x ds_load_b128
// (channel-contiguous records match the fragment's two 8-channel runs),
// reused by MT back-to-back WMMAs whose A fragments stream as b128 pairs.
//
// Pixel record stride = 40 halves (80B = 20 banks): conflict-reduced, 16B
// aligned. LDS: 4 waves * 54 px * 80B = 17.3KB/block. No barriers (wave-
// private tiles, in-order wave execution).
//
// blockDim = 128 (4 waves). grid.x = (B*H*W/16)/4, grid.y = CoutPad/(16*MT).
// ---------------------------------------------------------------------------
#define PXSTRIDE 40   // halves per pixel record (80 bytes)
#define WTILE    (54 * PXSTRIDE)

template <int MT>
__global__ __launch_bounds__(128) void conv3x3_wmma(
    const _Float16* __restrict__ x, const _Float16* __restrict__ wp,
    _Float16* __restrict__ yh, float* __restrict__ yf,
    const float* __restrict__ g, const float* __restrict__ bt,
    const float* __restrict__ mu, const float* __restrict__ vr,
    int Cin, int Cout, int H, int W, int CinPad, int last)
{
  __shared__ __align__(16) _Float16 smem[4 * WTILE];

  const int lane = threadIdx.x & 31;
  const int wave = threadIdx.x >> 5;
  const int n    = lane & 15;
  const int hi   = lane >> 4;

  const int stripsW   = W >> 4;
  const int stripsImg = H * stripsW;
  int strip = blockIdx.x * 4 + wave;
  int b   = strip / stripsImg;
  int s   = strip - b * stripsImg;
  int py  = s / stripsW;
  int px0 = (s - py * stripsW) << 4;

  const int HW  = H * W;
  const int ccn = CinPad >> 5;
  const int nCh = 9 * ccn;
  const _Float16* __restrict__ xb = x + (size_t)b * Cin * HW;
  _Float16* __restrict__ ldsw = smem + wave * WTILE;

  const int tileBase = blockIdx.y * MT;
  const _Float16* __restrict__ wlane =
      wp + (size_t)tileBase * nCh * 512 + (size_t)lane * 16;

  const bool colLok = (px0 > 0);
  const bool colRok = (px0 + 16 < W);
  const int  offL   = colLok ? -1 : 0;    // clamped, always-valid offsets
  const int  offR   = colRok ? 16 : 15;

  v8f acc[MT];
#pragma unroll
  for (int t = 0; t < MT; ++t) acc[t] = (v8f){};

  for (int cc = 0; cc < ccn; ++cc) {
    // ---- fill wave-private LDS tile for this 32-channel slab ----
    {
      int ch    = (cc << 5) + lane;
      bool chok = ch < Cin;
      const _Float16* plane = xb + (size_t)(chok ? ch : 0) * HW;
#pragma unroll
      for (int r = 0; r < 3; ++r) {
        int  yy  = py + r - 1;
        bool rok = (unsigned)yy < (unsigned)H;
        const _Float16* rowp = plane + (size_t)(rok ? yy : 0) * W + px0;
        // interior 16 pixels: two aligned 16B vector loads (one wait total)
        union { v8h v; _Float16 h[8]; } L0, L1;
        L0.v = *(const v8h*)(rowp);
        L1.v = *(const v8h*)(rowp + 8);
        _Float16 eL = rowp[offL];
        _Float16 eR = rowp[offR];
        bool mk  = rok & chok;
        bool mkL = mk & colLok;
        bool mkR = mk & colRok;
        _Float16* lp = ldsw + (r * 18) * PXSTRIDE + lane;
        lp[0]              = mkL ? eL : (_Float16)0.f;
        lp[17 * PXSTRIDE]  = mkR ? eR : (_Float16)0.f;
#pragma unroll
        for (int i = 0; i < 8; ++i)
          lp[(1 + i) * PXSTRIDE] = mk ? L0.h[i] : (_Float16)0.f;
#pragma unroll
        for (int i = 0; i < 8; ++i)
          lp[(9 + i) * PXSTRIDE] = mk ? L1.h[i] : (_Float16)0.f;
      }
    }

    // ---- 9 taps, one WMMA group each, all fed from the staged tile ----
#pragma unroll
    for (int tap = 0; tap < 9; ++tap) {
      const int chunk = cc * 9 + tap;
      // pixel index in tile: (dy+1)*18 + (dx+1) + n
      const _Float16* fp =
          ldsw + ((tap / 3) * 18 + (tap % 3) + n) * PXSTRIDE + (hi << 3);
      union { v16h v; v8h h2[2]; } Bf;
      Bf.h2[0] = *(const v8h*)fp;          // channels 8hi..8hi+7   (e=0..7)
      Bf.h2[1] = *(const v8h*)(fp + 16);   // channels 16+8hi..+7   (e=8..15)

#pragma unroll
      for (int t = 0; t < MT; ++t) {
        v16h A = *(const v16h*)(wlane + ((size_t)chunk + (size_t)t * nCh) * 512);
        acc[t] = __builtin_amdgcn_wmma_f32_16x16x32_f16(false, A, false, Bf.v,
                                                        (short)0, acc[t], false, false);
      }
    }
  }

  // C/D layout: VGPR r holds M = r + 8*hi, N = lane%16.
#pragma unroll
  for (int t = 0; t < MT; ++t) {
    union { v8f v; float f[8]; } R; R.v = acc[t];
#pragma unroll
    for (int r = 0; r < 8; ++r) {
      int co = ((tileBase + t) << 4) + r + (hi << 3);
      if (co < Cout) {
        float val = R.f[r];
        if (!last) {
          float inv = g[co] * rsqrtf(vr[co] + BN_EPS);
          val = val * inv + (bt[co] - mu[co] * inv);
        }
        val = fmaxf(val, 0.f);                       // ReLU on every layer (ref)
        size_t o = (((size_t)b * Cout + co) * H + py) * W + px0 + n;
        if (last) yf[o] = val; else yh[o] = (_Float16)val;
      }
    }
  }
}

// ---------------------------------------------------------------------------
// 2x2 maxpool with argmax (window order (0,0),(0,1),(1,0),(1,1); ties -> first)
// ---------------------------------------------------------------------------
__global__ void maxpool2x2(const _Float16* __restrict__ in, _Float16* __restrict__ out,
                           unsigned char* __restrict__ idx, int C, int H, int W, int B)
{
  int h = H >> 1, w = W >> 1;
  size_t total = (size_t)B * C * h * w;
  size_t i = (size_t)blockIdx.x * blockDim.x + threadIdx.x;
  if (i >= total) return;
  int xx = (int)(i % w); size_t t = i / w;
  int yy = (int)(t % h); t /= h;
  int c  = (int)(t % C);
  int b  = (int)(t / C);
  const _Float16* p = in + (((size_t)b * C + c) * H + 2 * yy) * W + 2 * xx;
  float v0 = (float)p[0], v1 = (float)p[1], v2 = (float)p[W], v3 = (float)p[W + 1];
  int bi = 0; float bv = v0;
  if (v1 > bv) { bv = v1; bi = 1; }
  if (v2 > bv) { bv = v2; bi = 2; }
  if (v3 > bv) { bv = v3; bi = 3; }
  out[i] = (_Float16)bv;
  idx[i] = (unsigned char)bi;
}

// ---------------------------------------------------------------------------
// 2x2 maxunpool, gather style.
// ---------------------------------------------------------------------------
__global__ void maxunpool2x2(const _Float16* __restrict__ in, const unsigned char* __restrict__ idx,
                             _Float16* __restrict__ out, int C, int h, int w, int B)
{
  int H = h * 2, Wd = w * 2;
  size_t total = (size_t)B * C * H * Wd;
  size_t i = (size_t)blockIdx.x * blockDim.x + threadIdx.x;
  if (i >= total) return;
  int xx = (int)(i % Wd); size_t t = i / Wd;
  int yy = (int)(t % H);  t /= H;
  int c  = (int)(t % C);
  int b  = (int)(t / C);
  size_t pi = (((size_t)b * C + c) * h + (yy >> 1)) * w + (xx >> 1);
  int sel = ((yy & 1) << 1) | (xx & 1);
  out[i] = (idx[pi] == sel) ? in[pi] : (_Float16)0.f;
}

// ---------------------------------------------------------------------------
// Channel softmax over 3 logit planes (B=4, HW=512*512).
// ---------------------------------------------------------------------------
__global__ void softmax3(const float* __restrict__ logits, float* __restrict__ out, int HW)
{
  size_t total = (size_t)4 * HW;
  size_t i = (size_t)blockIdx.x * blockDim.x + threadIdx.x;
  if (i >= total) return;
  int b = (int)(i / HW);
  int p = (int)(i % HW);
  const float* base = logits + (size_t)b * 3 * HW + p;
  float a0 = base[0], a1 = base[HW], a2 = base[2 * HW];
  float mx = fmaxf(a0, fmaxf(a1, a2));
  float e0 = __expf(a0 - mx), e1 = __expf(a1 - mx), e2 = __expf(a2 - mx);
  float inv = 1.f / (e0 + e1 + e2);
  float* ob = out + (size_t)b * 3 * HW + p;
  ob[0] = e0 * inv; ob[HW] = e1 * inv; ob[2 * HW] = e2 * inv;
}

// ---------------------------------------------------------------------------
// Host orchestration
// ---------------------------------------------------------------------------
static const int LAY[26][2] = {
  {3,16},{16,16},{16,32},{32,32},{32,64},{64,64},{64,64},
  {64,128},{128,128},{128,128},{128,128},{128,128},{128,128},
  {128,128},{128,128},{128,128},{128,128},{128,128},{128,64},
  {64,64},{64,64},{64,32},{32,32},{32,16},{16,16},{16,3}};

static void launch_conv(int MT, dim3 grid, hipStream_t stream,
                        const _Float16* x, const _Float16* wp,
                        _Float16* yh, float* yf,
                        const float* g, const float* bt, const float* mu, const float* vr,
                        int Cin, int Cout, int S, int CinPad, int last)
{
  if (MT == 4)
    conv3x3_wmma<4><<<grid, 128, 0, stream>>>(x, wp, yh, yf, g, bt, mu, vr,
                                              Cin, Cout, S, S, CinPad, last);
  else if (MT == 2)
    conv3x3_wmma<2><<<grid, 128, 0, stream>>>(x, wp, yh, yf, g, bt, mu, vr,
                                              Cin, Cout, S, S, CinPad, last);
  else
    conv3x3_wmma<1><<<grid, 128, 0, stream>>>(x, wp, yh, yf, g, bt, mu, vr,
                                              Cin, Cout, S, S, CinPad, last);
}

extern "C" void kernel_launch(void* const* d_in, const int* in_sizes, int n_in,
                              void* d_out, int out_size, void* d_ws, size_t ws_size,
                              hipStream_t stream) {
  (void)in_sizes; (void)n_in; (void)out_size; (void)ws_size;
  const int Bz = 4;
  const float* xin = (const float*)d_in[0];
  const float* Wt[26]; const float *G[25], *Bb[25], *Mu[25], *Vr[25];
  for (int i = 0; i < 26; ++i) Wt[i] = (const float*)d_in[1 + i];
  for (int i = 0; i < 25; ++i) {
    G[i]  = (const float*)d_in[27 + i];
    Bb[i] = (const float*)d_in[52 + i];
    Mu[i] = (const float*)d_in[77 + i];
    Vr[i] = (const float*)d_in[102 + i];
  }

  // Workspace carving (256B aligned).
  char* ws = (char*)d_ws;
  size_t off = 0;
  auto carve = [&](size_t bytes) -> char* {
    off = (off + 255) & ~(size_t)255;
    char* p = ws + off; off += bytes; return p;
  };
  const size_t actElems = (size_t)Bz * 16 * 512 * 512;   // largest f16 plane
  _Float16* actA = (_Float16*)carve(actElems * sizeof(_Float16));
  _Float16* actB = (_Float16*)carve(actElems * sizeof(_Float16));
  static const size_t idxElems[5] = {
    (size_t)4*16*256*256, (size_t)4*32*128*128, (size_t)4*64*64*64,
    (size_t)4*128*32*32,  (size_t)4*128*16*16 };
  unsigned char* idxP[5];
  for (int s = 0; s < 5; ++s) idxP[s] = (unsigned char*)carve(idxElems[s]);

  int CinP[26], CP[26];
  _Float16* Wp[26];
  for (int i = 0; i < 26; ++i) {
    int Cin = LAY[i][0], Cout = LAY[i][1];
    CinP[i] = (Cin + 31) / 32 * 32;
    CP[i]   = (Cout + 15) / 16 * 16;
    Wp[i] = (_Float16*)carve((size_t)CP[i] * 9 * CinP[i] * sizeof(_Float16));
  }

  // Pack all weights into fragment order (every call; deterministic).
  for (int i = 0; i < 26; ++i) {
    int total = (CP[i] >> 4) * (9 * (CinP[i] >> 5)) * 512;
    pack_weights<<<(total + 255) / 256, 256, 0, stream>>>(
        Wt[i], Wp[i], LAY[i][0], LAY[i][1], CinP[i], CP[i]);
  }

  // Convert input to f16.
  {
    size_t n = (size_t)Bz * 3 * 512 * 512;
    f32_to_f16<<<(unsigned)((n + 255) / 256), 256, 0, stream>>>(xin, actA, n);
  }

  _Float16* cur = actA;
  _Float16* nxt = actB;
  int S = 512;
  int slot = 0;
  float* logits = (float*)d_out;

  for (int i = 0; i < 26; ++i) {
    // Decoder unpool before selected layers.
    int us = -1;
    if (i == 13) us = 4; else if (i == 16) us = 3; else if (i == 19) us = 2;
    else if (i == 22) us = 1; else if (i == 24) us = 0;
    if (us >= 0) {
      int C = LAY[i][0];
      int h = S, w = S;
      S *= 2;
      size_t tot = (size_t)Bz * C * S * S;
      maxunpool2x2<<<(unsigned)((tot + 255) / 256), 256, 0, stream>>>(
          cur, idxP[us], nxt, C, h, w, Bz);
      _Float16* t = cur; cur = nxt; nxt = t;
    }

    int Cin = LAY[i][0], Cout = LAY[i][1];
    int last = (i == 25) ? 1 : 0;
    int MT = (Cout >= 64) ? 4 : (Cout == 32 ? 2 : 1);
    dim3 grid((unsigned)((size_t)Bz * S * S / 16 / 4),
              (unsigned)(CP[i] / (16 * MT)));
    launch_conv(MT, grid, stream, cur, Wp[i],
                last ? nullptr : nxt, last ? logits : nullptr,
                last ? nullptr : G[i], last ? nullptr : Bb[i],
                last ? nullptr : Mu[i], last ? nullptr : Vr[i],
                Cin, Cout, S, CinP[i], last);
    if (!last) { _Float16* t = cur; cur = nxt; nxt = t; }

    // Encoder pool after selected layers.
    if (i == 1 || i == 3 || i == 6 || i == 9 || i == 12) {
      size_t tot = (size_t)Bz * Cout * (S / 2) * (S / 2);
      maxpool2x2<<<(unsigned)((tot + 255) / 256), 256, 0, stream>>>(
          cur, nxt, idxP[slot], Cout, S, S, Bz);
      _Float16* t = cur; cur = nxt; nxt = t;
      S /= 2; ++slot;
    }
  }

  // Softmax over 3 channels; write after the logits region of d_out.
  {
    const int HW = 512 * 512;
    float* sm = logits + (size_t)Bz * 3 * HW;
    size_t tot = (size_t)Bz * HW;
    softmax3<<<(unsigned)((tot + 255) / 256), 256, 0, stream>>>(logits, sm, HW);
  }
}
